// MultiScaleGCN_87634512708198
// MI455X (gfx1250) — matrix operand order repaired and checked by reading, not compiled, not run on hardware
//
#include <hip/hip_runtime.h>
#include <hip/hip_bf16.h>

// Problem dims (fixed by reference)
#define BQ   16
#define NQ   2048
#define DQ   128      // hidden dim
#define KC   64       // K-chunk staged in LDS
#define MT   128      // rows per workgroup (8 waves x 16 rows)
#define SA   72       // LDS stride (elems) for A tile (row-major), 72*2B = 144B (16B-mult)
#define SB   72       // LDS stride for B tile (transposed: [col][k])
#define SH   136      // LDS stride for Hagg tile in phase 2 (136*2B = 272B, 16B-mult)
#define EPSQ 1e-5f

typedef unsigned short u16;
typedef __attribute__((ext_vector_type(16))) __bf16 v16bf;
typedef __attribute__((ext_vector_type(8)))  float  v8f;

union Frag16 { v16bf v; uint4 u[2]; };

// round-to-nearest-even f32 -> bf16 bits (pure integer, no scalar bf16 codegen needed)
__device__ __forceinline__ u16 f2bf(float f) {
    union { float f; unsigned u; } c; c.f = f;
    unsigned r = c.u + 0x7FFFu + ((c.u >> 16) & 1u);
    return (u16)(r >> 16);
}

// ---------------------------------------------------------------------------
// fp32 -> bf16 bulk convert (8 elems / thread / iter), used for x and adj
// ---------------------------------------------------------------------------
__global__ __launch_bounds__(256) void cvt_bf16_kernel(const float* __restrict__ src,
                                                       u16* __restrict__ dst, size_t n8) {
    size_t i = (size_t)blockIdx.x * blockDim.x + threadIdx.x;
    size_t stride = (size_t)gridDim.x * blockDim.x;
    for (; i < n8; i += stride) {
        const float4* s4 = (const float4*)(src + i * 8);
        float4 d0 = s4[0], d1 = s4[1];
        u16 e[8] = { f2bf(d0.x), f2bf(d0.y), f2bf(d0.z), f2bf(d0.w),
                     f2bf(d1.x), f2bf(d1.y), f2bf(d1.z), f2bf(d1.w) };
        *(uint4*)(dst + i * 8) = *(const uint4*)e;
    }
}

// W (128x128 row-major, [k][n]) -> transposed bf16 Wt[n][k] so that WMMA
// B-fragments are 16 contiguous K elements per lane.
__global__ __launch_bounds__(256) void cvt_w_kernel(const float* __restrict__ W,
                                                    u16* __restrict__ Wt) {
    int t = blockIdx.x * blockDim.x + threadIdx.x;   // 0..16383
    int k = t >> 7, n = t & 127;
    Wt[n * DQ + k] = f2bf(W[k * DQ + n]);
}

// ---------------------------------------------------------------------------
// Fused GCN layer: out = relu(LN( (adj @ H) @ W + b ) * g + beta)
//   grid = (N/MT, B), block = 256 (8 wave32)
// ---------------------------------------------------------------------------
template<bool ADJ_BF16, bool FINAL>
__global__ __launch_bounds__(256) void gcn_layer_kernel(
        const void* __restrict__ adjp,      // bf16[B][N][N] or f32[B][N][N]
        const u16*  __restrict__ hin,       // bf16[B][N][DQ]
        const u16*  __restrict__ Wt,        // bf16[DQ][DQ] transposed ([n][k])
        const float* __restrict__ bias,
        const float* __restrict__ gg,
        const float* __restrict__ bb_ln,
        u16*   __restrict__ hout,           // bf16 next-layer input (if !FINAL)
        float* __restrict__ fout)           // fp32 final output (if FINAL)
{
    __shared__ __align__(16) u16 smem[MT * SA + DQ * SB];   // 36,864 B; reused in phase 2
    u16* sA = smem;                // [MT][SA] adj tile, row-major over K
    u16* sB = smem + MT * SA;      // [DQ][SB] H tile transposed: [col][k]

    const int tid   = threadIdx.x;
    const int wave  = tid >> 5;
    const int lane  = tid & 31;
    const int lhalf = lane >> 4;        // 0: lanes 0-15, 1: lanes 16-31
    const int ln    = lane & 15;
    const int bb    = blockIdx.y;       // batch
    const int m0    = blockIdx.x * MT;  // row tile base

    // A(16x32) layout: lanes 0-15 hold K 0..15, lanes 16-31 hold K 16..31 (matches B)
    const int aK = lhalf * 16;
    // B(32x16) layout used here (same as A): lanes 0-15 hold K 0..15, lanes 16-31 hold K 16..31
    const int bK = lhalf * 16;

    v8f acc[8];
    #pragma unroll
    for (int cb = 0; cb < 8; ++cb)
        #pragma unroll
        for (int j = 0; j < 8; ++j) acc[cb][j] = 0.0f;

    // ---------------- GEMM1: Hagg(128x128 per WG) = adj_tile @ H ----------------
    for (int k0 = 0; k0 < NQ; k0 += KC) {
        __syncthreads();   // previous chunk's LDS reads complete

        // Stage A: adj rows m0..m0+127, cols k0..k0+63 (bf16) into sA
        if constexpr (ADJ_BF16) {
            const u16* agp = (const u16*)adjp + ((size_t)(bb * NQ + m0)) * NQ + k0;
            #pragma unroll
            for (int i = 0; i < 4; ++i) {
                int idx = tid + i * 256;           // 1024 x uint4
                int r = idx >> 3, c8 = idx & 7;
                uint4 d = *(const uint4*)(agp + (size_t)r * NQ + c8 * 8);
                *(uint4*)(&sA[r * SA + c8 * 8]) = d;
            }
            if (k0 + KC < NQ)   // L2 prefetch of next adj tile -> global_prefetch_b8
                __builtin_prefetch(agp + (size_t)(tid >> 1) * NQ + KC + (size_t)(tid & 1) * 32, 0, 0);
        } else {
            const float* agp = (const float*)adjp + ((size_t)(bb * NQ + m0)) * NQ + k0;
            #pragma unroll
            for (int i = 0; i < 4; ++i) {
                int idx = tid + i * 256;
                int r = idx >> 3, c8 = idx & 7;
                const float4* s4 = (const float4*)(agp + (size_t)r * NQ + c8 * 8);
                float4 d0 = s4[0], d1 = s4[1];
                u16 e[8] = { f2bf(d0.x), f2bf(d0.y), f2bf(d0.z), f2bf(d0.w),
                             f2bf(d1.x), f2bf(d1.y), f2bf(d1.z), f2bf(d1.w) };
                *(uint4*)(&sA[r * SA + c8 * 8]) = *(const uint4*)e;
            }
            if (k0 + KC < NQ)
                __builtin_prefetch(agp + (size_t)(tid >> 1) * NQ + KC + (size_t)(tid & 1) * 32, 0, 0);
        }

        // Stage B transposed: H rows k0..k0+63 (all 128 cols) -> sB[col][k]
        {
            const u16* hgp = hin + ((size_t)(bb * NQ + k0)) * DQ;
            #pragma unroll
            for (int i = 0; i < 4; ++i) {
                int idx = tid + i * 256;           // 1024 x uint4
                int k = idx >> 4, c8 = idx & 15;
                uint4 d = *(const uint4*)(hgp + (size_t)k * DQ + c8 * 8);
                const u16* e = (const u16*)&d;
                #pragma unroll
                for (int j = 0; j < 8; ++j) sB[(c8 * 8 + j) * SB + k] = e[j];
            }
        }
        __syncthreads();

        // 2 x (K=32) WMMA steps per chunk; 8 col-blocks per wave, A-frag reused 8x
        #pragma unroll
        for (int kk = 0; kk < KC; kk += 32) {
            Frag16 a;
            a.u[0] = *(const uint4*)(&sA[(wave * 16 + ln) * SA + kk + aK]);
            a.u[1] = *(const uint4*)(&sA[(wave * 16 + ln) * SA + kk + aK + 8]);
            #pragma unroll
            for (int cb = 0; cb < 8; ++cb) {
                Frag16 bf;
                bf.u[0] = *(const uint4*)(&sB[(cb * 16 + ln) * SB + kk + bK]);
                bf.u[1] = *(const uint4*)(&sB[(cb * 16 + ln) * SB + kk + bK + 8]);
                acc[cb] = __builtin_amdgcn_wmma_f32_16x16x32_bf16(
                    false, a.v, false, bf.v, (short)0, acc[cb], false, false);
            }
        }
    }

    // ---------------- GEMM2: Out(16x128 per wave) = Hagg_tile @ W ----------------
    __syncthreads();   // all GEMM1 LDS reads done; reuse smem as sH
    u16* sH = smem;    // [MT][SH] bf16 Hagg, row-major over K (=hidden dim)

    #pragma unroll
    for (int cb = 0; cb < 8; ++cb)
        #pragma unroll
        for (int j = 0; j < 8; ++j) {
            int rl = lhalf * 8 + j;   // C-layout: VGPR j -> row j (lanes 0-15) / j+8 (16-31)
            sH[(wave * 16 + rl) * SH + cb * 16 + ln] = f2bf(acc[cb][j]);
        }
    #pragma unroll
    for (int cb = 0; cb < 8; ++cb)
        #pragma unroll
        for (int j = 0; j < 8; ++j) acc[cb][j] = 0.0f;
    __syncthreads();

    #pragma unroll
    for (int kk = 0; kk < DQ; kk += 32) {
        Frag16 a;
        a.u[0] = *(const uint4*)(&sH[(wave * 16 + ln) * SH + kk + aK]);
        a.u[1] = *(const uint4*)(&sH[(wave * 16 + ln) * SH + kk + aK + 8]);
        #pragma unroll
        for (int cb = 0; cb < 8; ++cb) {
            Frag16 bf;   // W is tiny (32 KB bf16) and L2-hot: read B-frags straight from global
            bf.u[0] = *(const uint4*)(Wt + (cb * 16 + ln) * DQ + kk + bK);
            bf.u[1] = *(const uint4*)(Wt + (cb * 16 + ln) * DQ + kk + bK + 8);
            acc[cb] = __builtin_amdgcn_wmma_f32_16x16x32_bf16(
                false, a.v, false, bf.v, (short)0, acc[cb], false, false);
        }
    }

    // ---------------- bias + LayerNorm + ReLU + store ----------------
    float br[8], gr[8], ber[8];
    #pragma unroll
    for (int cb = 0; cb < 8; ++cb) {
        int c = cb * 16 + ln;
        br[cb] = bias[c]; gr[cb] = gg[c]; ber[cb] = bb_ln[c];
    }

    #pragma unroll
    for (int j = 0; j < 8; ++j) {
        float v[8], s = 0.f, q = 0.f;
        #pragma unroll
        for (int cb = 0; cb < 8; ++cb) {
            v[cb] = acc[cb][j] + br[cb];
            s += v[cb]; q += v[cb] * v[cb];
        }
        // each output row lives in one 16-lane half -> xor-reduce within the half
        #pragma unroll
        for (int m = 1; m < 16; m <<= 1) {
            s += __shfl_xor(s, m, 32);
            q += __shfl_xor(q, m, 32);
        }
        float mu   = s * (1.0f / 128.0f);
        float varr = q * (1.0f / 128.0f) - mu * mu;
        float rstd = rsqrtf(varr + EPSQ);
        int row = m0 + wave * 16 + lhalf * 8 + j;
        #pragma unroll
        for (int cb = 0; cb < 8; ++cb) {
            float o = fmaxf((v[cb] - mu) * rstd * gr[cb] + ber[cb], 0.0f);
            size_t off = ((size_t)(bb * NQ + row)) * DQ + cb * 16 + ln;
            if constexpr (FINAL) fout[off] = o;
            else                 hout[off] = f2bf(o);
        }
    }
}

// ---------------------------------------------------------------------------
extern "C" void kernel_launch(void* const* d_in, const int* in_sizes, int n_in,
                              void* d_out, int out_size, void* d_ws, size_t ws_size,
                              hipStream_t stream) {
    (void)in_sizes; (void)n_in; (void)out_size;
    const float* x   = (const float*)d_in[0];
    const float* adj = (const float*)d_in[1];
    const float* W[3]  = { (const float*)d_in[2], (const float*)d_in[6],  (const float*)d_in[10] };
    const float* bv[3] = { (const float*)d_in[3], (const float*)d_in[7],  (const float*)d_in[11] };
    const float* gv[3] = { (const float*)d_in[4], (const float*)d_in[8],  (const float*)d_in[12] };
    const float* ev[3] = { (const float*)d_in[5], (const float*)d_in[9],  (const float*)d_in[13] };

    const size_t hElems   = (size_t)BQ * NQ * DQ;      // 4,194,304
    const size_t adjElems = (size_t)BQ * NQ * NQ;      // 67,108,864

    u16* ws   = (u16*)d_ws;
    u16* h0   = ws;
    u16* h1   = h0 + hElems;
    u16* wt0  = h1 + hElems;
    u16* wt1  = wt0 + DQ * DQ;
    u16* wt2  = wt1 + DQ * DQ;
    u16* adjB = wt2 + DQ * DQ;
    const size_t needAdj = ((size_t)(adjB - ws) + adjElems) * sizeof(u16);
    const bool useAdjB = ws_size >= needAdj;   // 134 MB bf16 adj -> L2-resident across 3 layers

    // conversions
    cvt_bf16_kernel<<<2048, 256, 0, stream>>>(x, h0, hElems / 8);
    cvt_w_kernel<<<64, 256, 0, stream>>>(W[0], wt0);
    cvt_w_kernel<<<64, 256, 0, stream>>>(W[1], wt1);
    cvt_w_kernel<<<64, 256, 0, stream>>>(W[2], wt2);
    if (useAdjB)
        cvt_bf16_kernel<<<32768, 256, 0, stream>>>(adj, adjB, adjElems / 8);

    dim3 grid(NQ / MT, BQ);
    if (useAdjB) {
        gcn_layer_kernel<true,  false><<<grid, 256, 0, stream>>>(
            (const void*)adjB, h0, wt0, bv[0], gv[0], ev[0], h1, nullptr);
        gcn_layer_kernel<true,  false><<<grid, 256, 0, stream>>>(
            (const void*)adjB, h1, wt1, bv[1], gv[1], ev[1], h0, nullptr);
        gcn_layer_kernel<true,  true ><<<grid, 256, 0, stream>>>(
            (const void*)adjB, h0, wt2, bv[2], gv[2], ev[2], nullptr, (float*)d_out);
    } else {
        gcn_layer_kernel<false, false><<<grid, 256, 0, stream>>>(
            (const void*)adj,  h0, wt0, bv[0], gv[0], ev[0], h1, nullptr);
        gcn_layer_kernel<false, false><<<grid, 256, 0, stream>>>(
            (const void*)adj,  h1, wt1, bv[1], gv[1], ev[1], h0, nullptr);
        gcn_layer_kernel<false, true ><<<grid, 256, 0, stream>>>(
            (const void*)adj,  h0, wt2, bv[2], gv[2], ev[2], nullptr, (float*)d_out);
    }
}
